// CustomAttention_57784490001204
// MI455X (gfx1250) — matrix operand order repaired and checked by reading, not compiled, not run on hardware
//
#include <hip/hip_runtime.h>
#include <hip/hip_bf16.h>
#include <stdint.h>

// ---------------------------------------------------------------------------
// CustomAttention for MI455X (gfx1250, wave32)
//   q = x1 @ Wq.T ; k = x2 @ Wk.T ; v = x2 @ Wv.T   (proj kernel; v stored f16)
//   s[b,n,m] = mean_d tanh(q[b,n,d] + k[b,m,d])     (268M v_tanh_f32, q in regs)
//   (DropKey adds +/-1e-12 in fp32: below ulp of O(1) values -> exact no-op)
//   p = softmax_m(s) ; out = p @ v                  (v_wmma_f32_16x16x32_f16)
//   k/v tiles staged to LDS via GLOBAL_LOAD_ASYNC_TO_LDS_B128 (ASYNCcnt).
// ---------------------------------------------------------------------------

typedef __attribute__((ext_vector_type(16))) _Float16 v16h;
typedef __attribute__((ext_vector_type(8)))  float    v8f;

#define B_  16
#define N_  512
#define M_  512
#define D_  64

#if __has_builtin(__builtin_amdgcn_tanhf)
static __device__ __forceinline__ float fast_tanh(float x) { return __builtin_amdgcn_tanhf(x); }
#else
static __device__ __forceinline__ float fast_tanh(float x) { return tanhf(x); }
#endif

// 16-byte async copy global -> LDS (ASYNCcnt-tracked, no VGPR data path).
// lds_off: LDS byte address (low 32 bits of a generic LDS pointer).
static __device__ __forceinline__ void async_copy16(uint32_t lds_off, const void* gptr) {
    asm volatile("global_load_async_to_lds_b128 %0, %1, off"
                 :: "v"(lds_off), "v"(gptr) : "memory");
}

static __device__ __forceinline__ void wait_async0() {
#if __has_builtin(__builtin_amdgcn_s_wait_asynccnt)
    __builtin_amdgcn_s_wait_asynccnt(0);
#else
    asm volatile("s_wait_asynccnt 0" ::: "memory");
#endif
}

// ---------------------------------------------------------------------------
// Kernel 1: projections. blockIdx.y selects {q,k,v}; each block: 4 rows x 64 outs.
// v is written as f16 (only ever consumed as WMMA B-fragments).
// ---------------------------------------------------------------------------
__global__ __launch_bounds__(256) void ca_proj_kernel(
    const float* __restrict__ x1, const float* __restrict__ x2,
    const float* __restrict__ Wq, const float* __restrict__ Wk,
    const float* __restrict__ Wv,
    float* __restrict__ qo, float* __restrict__ ko, _Float16* __restrict__ vo)
{
    __shared__ float Ws[64 * 65];   // padded: bank = (o + d) % 64, conflict-free
    const int mat = blockIdx.y;
    const float* x = (mat == 0) ? x1 : x2;
    const float* W = (mat == 0) ? Wq : (mat == 1 ? Wk : Wv);

    const int tid = threadIdx.x;
    for (int i = tid; i < 64 * 64; i += 256) {
        int r = i >> 6, c = i & 63;
        Ws[r * 65 + c] = W[i];
    }
    __syncthreads();

    const int o_col = tid & 63;
    const int row   = blockIdx.x * 4 + (tid >> 6);     // 0..8191 = (b, n) flat
    const float* xr = x + row * D_;
    float acc = 0.f;
#pragma unroll
    for (int d = 0; d < D_; ++d) acc = fmaf(xr[d], Ws[o_col * 65 + d], acc);
    if (mat == 2) vo[row * D_ + o_col] = (_Float16)acc;
    else          ((mat == 0) ? qo : ko)[row * D_ + o_col] = acc;
}

// ---------------------------------------------------------------------------
// Kernel 2: fused scores + softmax + P@V.
// One workgroup (256 thr = 8 waves) per (batch, 16-row n-tile). Grid = 16*32.
// Dynamic LDS layout (bytes):
//   ss : 16 x 516 f32 scores                      @ 0     (33024)
//   pf : 16 x 520 f16 probabilities               @ 33024 (16640)
//   kv : union{ ks 128x68 f32 , vs 128x72 f16 }   @ 49664 (34816)
//   red: 4 x 32 x 8 f32 (K-split combine)         @ 84480 (4096)   total 88576
// ---------------------------------------------------------------------------
#define SS_STRIDE 516
#define PF_STRIDE 520
#define KS_STRIDE 68     // floats; 272 B row: 16B-aligned, conflict-free b128
#define VS_STRIDE 72     // halves; 144 B row: 16B-aligned
#define LDS_BYTES 88576

__global__ __launch_bounds__(256) void ca_attn_kernel(
    const float* __restrict__ qg, const float* __restrict__ kg,
    const _Float16* __restrict__ vg, float* __restrict__ out)
{
    extern __shared__ char smem[];
    float*    ss  = (float*)(smem);
    _Float16* pf  = (_Float16*)(smem + 33024);
    float*    ks  = (float*)(smem + 49664);
    _Float16* vs  = (_Float16*)(smem + 49664);
    float*    red = (float*)(smem + 84480);
    const uint32_t kv_lds = (uint32_t)(uintptr_t)(void*)(smem + 49664);

    const int tid  = threadIdx.x;
    const int lane = tid & 31;
    const int w    = tid >> 5;                 // wave 0..7
    const int bid  = blockIdx.x;
    const int b    = bid >> 5;                 // batch
    const int n0   = (bid & 31) * 16;          // n-tile base

    const float*    qb = qg + (size_t)b * N_ * D_;
    const float*    kb = kg + (size_t)b * M_ * D_;
    const _Float16* vb = vg + (size_t)b * M_ * D_;

    // ---- q row for this thread lives in registers (reused 2048x) ----
    const int sn = tid >> 4;       // 0..15 row within n-tile
    const int mb = tid & 15;       // m sub-column
    float4 qv[16];
    {
        const float4* qrow = (const float4*)(qb + (n0 + sn) * D_);
#pragma unroll
        for (int t = 0; t < 16; ++t) qv[t] = qrow[t];
    }

    // ---- scores: 4 m-tiles of 128 (k tile async-staged to LDS) ----
    for (int mt = 0; mt < 4; ++mt) {
        __syncthreads();           // previous ks readers done
        for (int i = tid; i < 2048; i += 256) {        // 128 rows x 16 float4
            const int r = i >> 4, j = i & 15;
            async_copy16(kv_lds + r * (KS_STRIDE * 4) + j * 16,
                         kb + (mt * 128 + r) * D_ + j * 4);
        }
        wait_async0();
        __syncthreads();
#pragma unroll 2
        for (int j = 0; j < 8; ++j) {
            const int m = mb + 16 * j;                 // 0..127
            const float4* kr = (const float4*)(ks + m * KS_STRIDE);
            float a = 0.f;
#pragma unroll
            for (int t = 0; t < 16; ++t) {
                const float4 kk = kr[t];
                a += fast_tanh(qv[t].x + kk.x);
                a += fast_tanh(qv[t].y + kk.y);
                a += fast_tanh(qv[t].z + kk.z);
                a += fast_tanh(qv[t].w + kk.w);
            }
            ss[sn * SS_STRIDE + mt * 128 + m] = a * 0.015625f;   // /64
        }
    }
    __syncthreads();

    // ---- softmax over m (each wave: 2 rows) ----
    for (int r = 0; r < 2; ++r) {
        const int n = w * 2 + r;
        const float* srow = ss + n * SS_STRIDE;
        float sv[16];
        float mx = -3.0e38f;
#pragma unroll
        for (int j = 0; j < 16; ++j) { sv[j] = srow[lane + 32 * j]; mx = fmaxf(mx, sv[j]); }
#pragma unroll
        for (int off = 16; off > 0; off >>= 1) mx = fmaxf(mx, __shfl_xor(mx, off, 32));
        float sum = 0.f;
#pragma unroll
        for (int j = 0; j < 16; ++j) { sv[j] = __expf(sv[j] - mx); sum += sv[j]; }
#pragma unroll
        for (int off = 16; off > 0; off >>= 1) sum += __shfl_xor(sum, off, 32);
        const float inv = 1.f / sum;
        _Float16* prow = pf + n * PF_STRIDE;
#pragma unroll
        for (int j = 0; j < 16; ++j) prow[lane + 32 * j] = (_Float16)(sv[j] * inv);
    }
    __syncthreads();   // pf ready; ks region free for vs

    // ---- P @ V via v_wmma_f32_16x16x32_f16 ----
    // wave -> (d-tile = w&3, K-half = w>>2); each m-tile of 128: 2 wmma / wave.
    const int dt  = w & 3;
    const int ksh = w >> 2;
    const int Ml  = lane & 15;
    const int hi  = lane >> 4;
    v8f acc = {0.f, 0.f, 0.f, 0.f, 0.f, 0.f, 0.f, 0.f};

    for (int mt = 0; mt < 4; ++mt) {
        for (int i = tid; i < 1024; i += 256) {        // 128 rows x 8 halves/xfer
            const int r = i >> 3, j = i & 7;
            async_copy16(kv_lds + r * (VS_STRIDE * 2) + j * 16,
                         vb + (mt * 128 + r) * D_ + j * 8);
        }
        wait_async0();
        __syncthreads();
#pragma unroll
        for (int kk = 0; kk < 2; ++kk) {
            const int klocal = ksh * 64 + kk * 32;        // K base within m-tile
            const int mglob  = mt * 128 + klocal;         // global m base
            // A fragment (16x32 f16): lanes 0-15 hold K {0..7,16..23}, 16-31 {8..15,24..31}
            union { v16h h; uint32_t u[8]; } A;
            const uint32_t* ap =
                (const uint32_t*)pf + ((Ml * PF_STRIDE + mglob + hi * 8) >> 1);
#pragma unroll
            for (int t = 0; t < 4; ++t) { A.u[t] = ap[t]; A.u[4 + t] = ap[8 + t]; }
            // B fragment (32x16 f16): lane = column d; lanes 0-15 K {0..7,16..23}, 16-31 {8..15,24..31}
            v16h Bf;
#pragma unroll
            for (int t = 0; t < 16; ++t)
                Bf[t] = vs[(klocal + hi * 8 + (t & 7) + (t >> 3) * 16) * VS_STRIDE + dt * 16 + Ml];
            acc = __builtin_amdgcn_wmma_f32_16x16x32_f16(
                false, A.h, false, Bf, (short)0, acc, false, false);
        }
        __syncthreads();   // vs consumed; safe to overwrite next iter
    }

    // ---- combine K-split halves and store ----
    if (ksh == 1) {
        float* rr = red + ((w - 4) * 32 + lane) * 8;
#pragma unroll
        for (int j = 0; j < 8; ++j) rr[j] = acc[j];
    }
    __syncthreads();
    if (ksh == 0) {
        const float* rr = red + (w * 32 + lane) * 8;
        const int col = dt * 16 + Ml;
#pragma unroll
        for (int j = 0; j < 8; ++j) {
            const int row = j + 8 * hi;    // C/D layout: VGPR j -> M = j (+8 for hi lanes)
            out[((size_t)b * N_ + (n0 + row)) * D_ + col] = acc[j] + rr[j];
        }
    }
}

// ---------------------------------------------------------------------------
extern "C" void kernel_launch(void* const* d_in, const int* in_sizes, int n_in,
                              void* d_out, int out_size, void* d_ws, size_t ws_size,
                              hipStream_t stream) {
    (void)in_sizes; (void)n_in; (void)out_size; (void)ws_size;
    const float* input1 = (const float*)d_in[0];
    const float* input2 = (const float*)d_in[1];
    const float* Wq     = (const float*)d_in[2];
    const float* Wk     = (const float*)d_in[3];
    const float* Wv     = (const float*)d_in[4];

    float*     q  = (float*)d_ws;                          // 2 MB f32
    float*     k  = q + (size_t)B_ * N_ * D_;              // 2 MB f32
    _Float16*  vh = (_Float16*)((char*)d_ws + 4u * 1024 * 1024);  // 1 MB f16

    ca_proj_kernel<<<dim3(2048, 3), 256, 0, stream>>>(
        input1, input2, Wq, Wk, Wv, q, k, vh);

    ca_attn_kernel<<<dim3(B_ * (N_ / 16)), 256, LDS_BYTES, stream>>>(
        q, k, vh, (float*)d_out);
}
